// GAT_29033978921226
// MI455X (gfx1250) — matrix operand order repaired and checked by reading, not compiled, not run on hardware
//
#include <hip/hip_runtime.h>
#include <hip/hip_bf16.h>

// ---- CDNA5 WMMA fragment types (wave32) ----
typedef __attribute__((ext_vector_type(16))) _Float16 v16h;
typedef __attribute__((ext_vector_type(8)))  _Float16 v8h;
typedef __attribute__((ext_vector_type(8)))  float    v8f;

#define GAT_N   14
#define GAT_E   182
#define GAT_ET  196          // E + N self loops
#define GAT_H   4
#define GAT_F   128
#define GAT_C   64
#define HC      256          // H*C
#define NEG_SLOPE 0.2f

// -------- prologue: transpose + fp16-convert the GEMM weights into d_ws --------
// w1t: [256][128] fp16  (W1 is [128,256] fp32, K-major per output column after transpose)
// w2t: [256][64]  fp16
__global__ void gat_wconv(const float* __restrict__ W1, const float* __restrict__ W2,
                          _Float16* __restrict__ w1t, _Float16* __restrict__ w2t) {
  int i = blockIdx.x * 256 + threadIdx.x;
  if (i < HC * GAT_F) { int n = i >> 7, k = i & 127; w1t[i] = (_Float16)W1[k * HC + n]; }
  if (i < HC * GAT_C) { int n = i >> 6, k = i & 63;  w2t[i] = (_Float16)W2[k * HC + n]; }
}

// -------- [16,K] (LDS fp16) x [K,256] (global fp16, stored [256][K]) -> [16,256] fp32 LDS --------
// 8 waves per block; wave w owns output N-tiles {w, w+8}; K/32 WMMA steps each.
template <int K>
__device__ __forceinline__ void gat_gemm(const _Float16* xLDS,
                                         const _Float16* __restrict__ Wt,
                                         float* outLDS) {
  const int lane = threadIdx.x & 31;
  const int wid  = threadIdx.x >> 5;
  const int ksel = lane >> 4;        // 0: lanes 0-15, 1: lanes 16-31
  const int m    = lane & 15;
  v8f acc0 = {}; v8f acc1 = {};
  const _Float16* b0p = Wt + (size_t)(wid * 16 + m) * K;        // N-tile wid
  const _Float16* b1p = Wt + (size_t)((wid + 8) * 16 + m) * K;  // N-tile wid+8
#pragma unroll
  for (int kb = 0; kb < K; kb += 32) {
    // A 16x32 fp16 layout: lane<16 holds K {kb..kb+7, kb+16..kb+23}, lane>=16 the other halves
    const int k0 = kb + ksel * 8;
    v8h lo = *(const v8h*)&xLDS[m * K + k0];
    v8h hi = *(const v8h*)&xLDS[m * K + k0 + 16];
    v16h a = __builtin_shufflevector(lo, hi, 0,1,2,3,4,5,6,7,8,9,10,11,12,13,14,15);
    // B 32x16 fp16 layout: same K order as A (lane<16: kb..kb+7, kb+16..kb+23; lane>=16 the other halves)
    const int kb2 = k0;
    v16h b0 = __builtin_shufflevector(*(const v8h*)&b0p[kb2], *(const v8h*)&b0p[kb2 + 16], 0,1,2,3,4,5,6,7,8,9,10,11,12,13,14,15);
    v16h b1 = __builtin_shufflevector(*(const v8h*)&b1p[kb2], *(const v8h*)&b1p[kb2 + 16], 0,1,2,3,4,5,6,7,8,9,10,11,12,13,14,15);
    acc0 = __builtin_amdgcn_wmma_f32_16x16x32_f16(false, a, false, b0, (short)0, acc0, false, false);
    acc1 = __builtin_amdgcn_wmma_f32_16x16x32_f16(false, a, false, b1, (short)0, acc1, false, false);
  }
  // C/D layout: VGPR r -> (M = r + 8*ksel, N = lane&15)
  const int row = ksel * 8;
  const int c0  = wid * 16 + m;
#pragma unroll
  for (int r = 0; r < 8; ++r) {
    outLDS[(row + r) * HC + c0]       = acc0[r];
    outLDS[(row + r) * HC + c0 + 128] = acc1[r];
  }
}

// -------- GAT attention: logits, leaky-relu, dst-segment softmax, CSR aggregate --------
__device__ __forceinline__ void gat_attention(
    const float* hbuf, const int* esrc, const int* edst,
    const float* __restrict__ a_src, const float* __restrict__ a_dst,
    float* eS, float* alS, float* alD,
    int* cnt, int* off, int* cur, int* order, float* agg) {
  const int t = threadIdx.x;
  // per-(node, head) attention logits: al = <h[n,h,:], a[h,:]>
  if (t < 64) {
    int n = t >> 2, h = t & 3;
    float s1 = 0.f, s2 = 0.f;
    const float* hp = &hbuf[n * HC + h * GAT_C];
    for (int c = 0; c < GAT_C; ++c) {
      float hv = hp[c];
      s1 += hv * a_src[h * GAT_C + c];
      s2 += hv * a_dst[h * GAT_C + c];
    }
    alS[t] = s1; alD[t] = s2;
  }
  // per-dst edge counts (CSR)
  if (t >= 224 && t < 240) {
    int n = t - 224, c = 0;
    for (int e = 0; e < GAT_ET; ++e) c += (edst[e] == n);
    cnt[n] = c;
  }
  __syncthreads();
  // edge logits with leaky relu
  if (t < GAT_ET) {
    int s = esrc[t], d = edst[t];
#pragma unroll
    for (int h = 0; h < GAT_H; ++h) {
      float v = alS[s * 4 + h] + alD[d * 4 + h];
      eS[t * 4 + h] = (v > 0.f) ? v : NEG_SLOPE * v;
    }
  }
  if (t == 208) {          // CSR prefix sum (tiny, single lane)
    int o = 0;
    for (int n = 0; n < 16; ++n) { off[n] = o; cur[n] = o; o += cnt[n]; }
  }
  __syncthreads();
  // stable softmax per (dst, head): each owned by one thread, disjoint writes
  if (t < GAT_N * GAT_H) {
    int n = t >> 2, h = t & 3;
    float m = -3.0e38f;
    for (int e = 0; e < GAT_ET; ++e) if (edst[e] == n) m = fmaxf(m, eS[e * 4 + h]);
    float s = 0.f;
    for (int e = 0; e < GAT_ET; ++e) if (edst[e] == n) {
      float v = __expf(eS[e * 4 + h] - m); eS[e * 4 + h] = v; s += v;
    }
    float inv = 1.f / s;   // every dst has a self loop -> s > 0
    for (int e = 0; e < GAT_ET; ++e) if (edst[e] == n) eS[e * 4 + h] *= inv;
  }
  __syncthreads();
  // bucket edges by dst
  if (t < GAT_ET) { int p = atomicAdd(&cur[edst[t]], 1); order[p] = t; }
  __syncthreads();
  // aggregate: thread owns (head, channel); loop dst nodes and their incident edges
  {
    int h = t >> 6, c = t & 63;
    for (int n = 0; n < GAT_N; ++n) {
      float acc = 0.f;
      int b = off[n], e2 = b + cnt[n];
      for (int j = b; j < e2; ++j) {
        int ed = order[j];
        acc += eS[ed * 4 + h] * hbuf[esrc[ed] * HC + h * GAT_C + c];
      }
      agg[(n * 4 + h) * GAT_C + c] = acc;
    }
  }
  __syncthreads();
}

// -------- fully fused GAT forward: one workgroup (8 wave32s) per graph --------
__global__ __launch_bounds__(256) void gat_fused(
    const float* __restrict__ feature, const int* __restrict__ edge_list,
    const _Float16* __restrict__ w1t, const _Float16* __restrict__ w2t,
    const float* __restrict__ a_src1, const float* __restrict__ a_dst1, const float* __restrict__ b1,
    const float* __restrict__ a_src2, const float* __restrict__ a_dst2, const float* __restrict__ b2,
    const float* __restrict__ W_lin, const float* __restrict__ b_lin,
    const float* __restrict__ W_pred, const float* __restrict__ b_pred,
    float* __restrict__ out) {
  __shared__ __align__(32) _Float16 xh[16 * GAT_F];     // padded fp16 node features
  __shared__ float hbuf[16 * HC];                       // GEMM output [16][H*C]
  __shared__ float eS[GAT_ET * GAT_H];                  // edge logits -> alphas
  __shared__ int esrc[GAT_ET], edst[GAT_ET];
  __shared__ float alS[64], alD[64];
  __shared__ int cnt[16], off[16], cur[16], order[GAT_ET];
  __shared__ float agg[GAT_N * GAT_H * GAT_C];
  __shared__ __align__(32) _Float16 x2h[16 * GAT_C];
  __shared__ float h2f[GAT_N * GAT_C];
  __shared__ float zred[8 * 32];
  __shared__ float zfin[32];

  const int g = blockIdx.x;
  const int t = threadIdx.x;

  // stage features (rows 14,15 zero-padded) and edges (+ self loops)
  const float* xg = feature + (size_t)g * GAT_N * GAT_F;
  for (int i = t; i < 16 * GAT_F; i += 256) {
    int n = i >> 7;
    xh[i] = (_Float16)((n < GAT_N) ? xg[i] : 0.f);
  }
  const int* eg = edge_list + (size_t)g * GAT_E * 2;
  if (t < GAT_ET) {
    int s, d;
    if (t < GAT_E) { s = eg[2 * t]; d = eg[2 * t + 1]; } else { s = d = t - GAT_E; }
    esrc[t] = s; edst[t] = d;
  }
  __syncthreads();

  // ---- layer 1 ----
  gat_gemm<GAT_F>(xh, w1t, hbuf);
  __syncthreads();
  gat_attention(hbuf, esrc, edst, a_src1, a_dst1, eS, alS, alD, cnt, off, cur, order, agg);
  for (int i = t; i < 16 * GAT_C; i += 256) {   // mean over heads + bias + ReLU -> fp16 input of layer 2
    int n = i >> 6, c = i & 63;
    float v = 0.f;
    if (n < GAT_N) {
      v = 0.25f * (agg[(n * 4 + 0) * 64 + c] + agg[(n * 4 + 1) * 64 + c] +
                   agg[(n * 4 + 2) * 64 + c] + agg[(n * 4 + 3) * 64 + c]) + b1[c];
      v = fmaxf(v, 0.f);
    }
    x2h[i] = (_Float16)v;
  }
  __syncthreads();

  // ---- layer 2 ----
  gat_gemm<GAT_C>(x2h, w2t, hbuf);
  __syncthreads();
  gat_attention(hbuf, esrc, edst, a_src2, a_dst2, eS, alS, alD, cnt, off, cur, order, agg);
  for (int i = t; i < GAT_N * GAT_C; i += 256) { // mean over heads + bias (no ReLU)
    int n = i >> 6, c = i & 63;
    h2f[i] = 0.25f * (agg[(n * 4 + 0) * 64 + c] + agg[(n * 4 + 1) * 64 + c] +
                      agg[(n * 4 + 2) * 64 + c] + agg[(n * 4 + 3) * 64 + c]) + b2[c];
  }
  __syncthreads();

  // ---- readout: z = flat(h2f)[896] @ W_lin[896,32] + b_lin; p = z @ W_pred + b_pred; sigmoid ----
  {
    int j = t & 31, ch = t >> 5;
    float p = 0.f;
    const int i0 = ch * 112;                    // 8 chunks x 112 = 896
    for (int i = i0; i < i0 + 112; ++i) p += h2f[i] * W_lin[i * 32 + j];
    zred[t] = p;
  }
  __syncthreads();
  if (t < 32) {
    float z = b_lin[t];
    for (int ch = 0; ch < 8; ++ch) z += zred[ch * 32 + t];
    zfin[t] = z;
  }
  __syncthreads();
  if (t == 0) {
    float p = b_pred[0];
    for (int j = 0; j < 32; ++j) p += zfin[j] * W_pred[j];
    out[g] = 1.f / (1.f + __expf(-p));
  }
}

extern "C" void kernel_launch(void* const* d_in, const int* in_sizes, int n_in,
                              void* d_out, int out_size, void* d_ws, size_t ws_size,
                              hipStream_t stream) {
  const float* feature   = (const float*)d_in[0];
  const int*   edge_list = (const int*)d_in[1];   // integer input -> const int* per harness contract
  const float* W1        = (const float*)d_in[2];
  const float* a_src1    = (const float*)d_in[3];
  const float* a_dst1    = (const float*)d_in[4];
  const float* b1        = (const float*)d_in[5];
  const float* W2        = (const float*)d_in[6];
  const float* a_src2    = (const float*)d_in[7];
  const float* a_dst2    = (const float*)d_in[8];
  const float* b2        = (const float*)d_in[9];
  const float* W_lin     = (const float*)d_in[10];
  const float* b_lin     = (const float*)d_in[11];
  const float* W_pred    = (const float*)d_in[12];
  const float* b_pred    = (const float*)d_in[13];
  float* out = (float*)d_out;

  // workspace: w1t fp16 [256][128] (64 KB) + w2t fp16 [256][64] (32 KB)
  _Float16* w1t = (_Float16*)d_ws;
  _Float16* w2t = (_Float16*)((char*)d_ws + (size_t)HC * GAT_F * sizeof(_Float16));

  gat_wconv<<<(HC * GAT_F + 255) / 256, 256, 0, stream>>>(W1, W2, w1t, w2t);

  const int B = in_sizes[0] / (GAT_N * GAT_F);   // 16384
  gat_fused<<<B, 256, 0, stream>>>(feature, edge_list, w1t, w2t,
                                   a_src1, a_dst1, b1, a_src2, a_dst2, b2,
                                   W_lin, b_lin, W_pred, b_pred, out);
}
